// ObjectGNN_68118181314662
// MI455X (gfx1250) — compile-verified, matrix operand order checked by reading, not run on hardware
//
#include <hip/hip_runtime.h>

typedef __attribute__((ext_vector_type(16))) __bf16 v16bf;
typedef __attribute__((ext_vector_type(8)))  float  v8f;

#define NDIM 16
#define HDIM 64
#define MAXO 20
#define NL   3
#define OUTC 9900
#define GPB  16
#define ROWS (GPB*MAXO)          // 320 rows per block

// --- LDS strides (bytes), padded to avoid bank conflicts, 16B aligned ---
#define H_SB   272               // h rows: 64 f32 (+pad)
#define A_SB   272               // concat rows: 128 bf16 / 64 f32 tmp (+pad)
#define ACT_SB 144               // act rows: 64 bf16 (+pad)
#define W1_SB  272               // w1^T rows: 128 bf16 (+pad)
#define W2_SB  144               // w2^T rows: 64 bf16 (+pad)

// --- LDS layout for fused GNN kernel ---
#define OFF_H    0
#define OFF_A    (OFF_H   + ROWS*H_SB)     // 87040
#define OFF_ACT  (OFF_A   + ROWS*A_SB)     // 174080
#define OFF_W1   (OFF_ACT + ROWS*ACT_SB)   // 220160
#define OFF_W2   (OFF_W1  + 64*W1_SB)      // 237568
#define OFF_B1   (OFF_W2  + 64*W2_SB)      // 246784
#define OFF_B2   (OFF_B1  + 256)           // 247040
#define OFF_MEAN (OFF_B2  + 256)           // 247296 (16x64 f32; also ln_g/ln_b staging)
#define OFF_NN   (OFF_MEAN + 4096)         // 251392
#define SMEM_K1  (OFF_NN  + 64)            // 251456 bytes (< 320KB/WGP)

#define HPTR(r) ((float*)(sm + OFF_H + (size_t)(r)*H_SB))

// native bf16 convert (RNE) -> lets clang emit hardware cvt ops
__device__ __forceinline__ void st_bf16(char* p, float x) {
  *(__bf16*)p = (__bf16)x;
}

// Load one 16x32 bf16 WMMA fragment from row-major LDS (row stride in bytes).
// Per CDNA5 ISA 16-bit A/B layout: lanes 0-15 hold K {kk..kk+7, kk+16..kk+23},
// lanes 16-31 hold K {kk+8..kk+15, kk+24..kk+31}, row/col = rc0 + (lane&15).
__device__ __forceinline__ v16bf ld_frag(const char* base, int rc0, int strideB, int kk) {
  int lane = (int)(threadIdx.x & 31u);
  const char* p = base + (rc0 + (lane & 15)) * strideB + (kk + (lane >> 4) * 8) * 2;
  union { v16bf v; float4 f[2]; } u;
  u.f[0] = *(const float4*)(p);
  u.f[1] = *(const float4*)(p + 32);
  return u.v;
}

#define WMMA_BF16(a, b, c) \
  __builtin_amdgcn_wmma_f32_16x16x32_bf16(false, (a), false, (b), (short)0, (c), false, false)

// Block-level GEMM over 320xKS @ KSx64 with 8 waves, 80 tiles.
// Two tiles per iteration with independent accumulators to overlap the
// WMMA->WMMA RAW hazard window (5 NOPs for bf16) with the sibling chain.
// MODE 0: relu(x+bias) -> bf16 act buffer      (dst stride ACT_SB)
// MODE 1: x+bias       -> bf16 into A cols 64+ (dst stride A_SB, +128B)
// MODE 2: x+bias       -> f32 tmp              (dst stride A_SB)
template<int KS, int MODE>
__device__ __forceinline__ void block_gemm(const char* A, int aStride,
                                           const char* W, int wStride,
                                           const float* bias, char* dst) {
  int wave = (int)(threadIdx.x >> 5), lane = (int)(threadIdx.x & 31u);
  int col16 = lane & 15, half = lane >> 4;
  for (int s = 0; s < 5; ++s) {
    int t0 = wave * 10 + 2 * s, t1 = t0 + 1;
    int m0 = (t0 >> 2) * 16, n0 = (t0 & 3) * 16;
    int m1 = (t1 >> 2) * 16, n1 = (t1 & 3) * 16;
    v8f acc0 = {0.f,0.f,0.f,0.f,0.f,0.f,0.f,0.f};
    v8f acc1 = {0.f,0.f,0.f,0.f,0.f,0.f,0.f,0.f};
#pragma unroll
    for (int kk = 0; kk < KS; kk += 32) {
      v16bf a0 = ld_frag(A, m0, aStride, kk);
      v16bf b0 = ld_frag(W, n0, wStride, kk);
      v16bf a1 = ld_frag(A, m1, aStride, kk);
      v16bf b1 = ld_frag(W, n1, wStride, kk);
      acc0 = WMMA_BF16(a0, b0, acc0);
      acc1 = WMMA_BF16(a1, b1, acc1);
    }
#pragma unroll
    for (int q = 0; q < 2; ++q) {
      int mt = q ? m1 : m0, nt = q ? n1 : n0;
      v8f& acc = q ? acc1 : acc0;
      int col = nt + col16;
      float bs = bias[col];
#pragma unroll
      for (int v = 0; v < 8; ++v) {
        int row = mt + v + half * 8;
        float x = acc[v] + bs;
        if (MODE == 0) {
          st_bf16(dst + row*ACT_SB + col*2, fmaxf(x, 0.f));
        } else if (MODE == 1) {
          st_bf16(dst + row*A_SB + 128 + col*2, x);
        } else {
          *(float*)(dst + row*A_SB + col*4) = x;
        }
      }
    }
  }
}

// Stage one layer's weights transposed (w^T[n][k]) as bf16 + biases into LDS.
__device__ __forceinline__ void load_weights(char* sm, const float* w1, const float* b1,
                                             const float* w2, const float* b2, int tid) {
  for (int e = tid; e < 128*64; e += 256) {      // w1: [128][64] -> w1t[n][k]
    int k = e >> 6, n = e & 63;
    st_bf16(sm + OFF_W1 + n*W1_SB + k*2, w1[e]);
  }
  for (int e = tid; e < 64*64; e += 256) {       // w2: [64][64] -> w2t[n][k]
    int k = e >> 6, n = e & 63;
    st_bf16(sm + OFF_W2 + n*W2_SB + k*2, w2[e]);
  }
  if (tid < 64) {
    ((float*)(sm + OFF_B1))[tid] = b1[tid];
    ((float*)(sm + OFF_B2))[tid] = b2[tid];
  }
}

__global__ __launch_bounds__(256) void gnn_fused(
    const float* __restrict__ nf, const int* __restrict__ nnodes,
    const float* __restrict__ embed_w, const float* __restrict__ embed_b,
    const float* __restrict__ edge_w1, const float* __restrict__ edge_b1,
    const float* __restrict__ edge_w2, const float* __restrict__ edge_b2,
    const float* __restrict__ node_w1, const float* __restrict__ node_b1,
    const float* __restrict__ node_w2, const float* __restrict__ node_b2,
    const float* __restrict__ ln_g, const float* __restrict__ ln_b,
    float* __restrict__ gv_out) {
  extern __shared__ char sm[];
  const int tid = (int)threadIdx.x;
  const int g0 = (int)blockIdx.x * GPB;

  int*   nnS   = (int*)(sm + OFF_NN);
  float* b1S   = (float*)(sm + OFF_B1);
  float* b2S   = (float*)(sm + OFF_B2);
  float* meanS = (float*)(sm + OFF_MEAN);

  // ---- stage inputs: node features (in A region), embed weights ----
  if (tid < GPB) nnS[tid] = nnodes[g0 + tid];
  float* nfS = (float*)(sm + OFF_A);                    // [320][16] f32
  for (int e = tid; e < ROWS*NDIM; e += 256)
    nfS[e] = nf[(size_t)g0 * MAXO * NDIM + e];
  float* ewS = (float*)(sm + OFF_W1);                   // [16][64] f32
  for (int e = tid; e < NDIM*HDIM; e += 256) ewS[e] = embed_w[e];
  if (tid < HDIM) b1S[tid] = embed_b[tid];
  __syncthreads();

  // ---- embed: h = nf @ embed_w + b (K=16, VALU) ----
  for (int e = tid; e < ROWS*HDIM; e += 256) {
    int r = e >> 6, c = e & 63;
    float acc = b1S[c];
#pragma unroll
    for (int d = 0; d < NDIM; ++d) acc += nfS[r*NDIM + d] * ewS[d*HDIM + c];
    HPTR(r)[c] = acc;
  }
  __syncthreads();

  for (int l = 0; l < NL; ++l) {
    // ---- masked mean per graph ----
    for (int e = tid; e < GPB*HDIM; e += 256) {
      int g = e >> 6, c = e & 63, n = nnS[g];
      float s = 0.f;
      for (int i = 0; i < n; ++i) s += HPTR(g*MAXO + i)[c];
      meanS[e] = s / (float)n;
    }
    __syncthreads();
    // ---- build A = [h | mean] bf16 ----
    for (int e = tid; e < ROWS*128; e += 256) {
      int r = e >> 7, k = e & 127;
      float x = (k < 64) ? HPTR(r)[k] : meanS[(r / MAXO) * HDIM + (k - 64)];
      st_bf16(sm + OFF_A + r*A_SB + k*2, x);
    }
    __syncthreads();
    // ---- edge MLP ----
    load_weights(sm, edge_w1 + l*128*64, edge_b1 + l*64, edge_w2 + l*64*64, edge_b2 + l*64, tid);
    __syncthreads();
    block_gemm<128,0>(sm+OFF_A, A_SB, sm+OFF_W1, W1_SB, b1S, sm+OFF_ACT);
    __syncthreads();
    block_gemm<64,1>(sm+OFF_ACT, ACT_SB, sm+OFF_W2, W2_SB, b2S, sm+OFF_A); // edge_msg -> A[:,64:]
    __syncthreads();
    // ---- node MLP (A already holds [h | edge_msg]) ----
    load_weights(sm, node_w1 + l*128*64, node_b1 + l*64, node_w2 + l*64*64, node_b2 + l*64, tid);
    if (tid < 64)            meanS[tid] = ln_g[l*64 + tid];        // stage LN params
    else if (tid < 128)      meanS[tid] = ln_b[l*64 + tid - 64];
    __syncthreads();
    block_gemm<128,0>(sm+OFF_A, A_SB, sm+OFF_W1, W1_SB, b1S, sm+OFF_ACT);
    __syncthreads();
    block_gemm<64,2>(sm+OFF_ACT, ACT_SB, sm+OFF_W2, W2_SB, b2S, sm+OFF_A); // f32 tmp overlays A
    __syncthreads();
    // ---- residual + LayerNorm + mask ----
    for (int r = tid; r < ROWS; r += 256) {
      int g = r / MAXO, i = r - g*MAXO;
      float mk = (i < nnS[g]) ? 1.f : 0.f;
      float xv[HDIM];
      float m = 0.f;
#pragma unroll
      for (int c = 0; c < HDIM; ++c) {
        xv[c] = HPTR(r)[c] + *(const float*)(sm + OFF_A + r*A_SB + c*4);
        m += xv[c];
      }
      m *= (1.f/64.f);
      float var = 0.f;
#pragma unroll
      for (int c = 0; c < HDIM; ++c) { float d = xv[c] - m; var += d*d; }
      var *= (1.f/64.f);
      float rinv = rsqrtf(var + 1e-5f);
#pragma unroll
      for (int c = 0; c < HDIM; ++c)
        HPTR(r)[c] = ((xv[c] - m) * rinv * meanS[c] + meanS[64 + c]) * mk;
    }
    __syncthreads();
  }
  // ---- graph vector ----
  for (int e = tid; e < GPB*HDIM; e += 256) {
    int g = e >> 6, c = e & 63, n = nnS[g];
    float s = 0.f;
    for (int i = 0; i < n; ++i) s += HPTR(g*MAXO + i)[c];
    gv_out[(size_t)(g0 + g)*HDIM + c] = s / (float)n;
  }
}

// ---- readout stage 1: t = relu(gv @ ro_w1 + b1), bf16 out ----
#define K2_OFF_A 0                 // 128*144
#define K2_OFF_W 18432             // 64*144
#define K2_OFF_B 27648
#define SMEM_K2A 27904

__global__ __launch_bounds__(256) void readout_stage1(
    const float* __restrict__ gv, const float* __restrict__ w1,
    const float* __restrict__ b1, __bf16* __restrict__ tout) {
  extern __shared__ char sm[];
  int tid = (int)threadIdx.x;
  size_t r0 = (size_t)blockIdx.x * 128;
  for (int e = tid; e < 128*64; e += 256) {
    int r = e >> 6, c = e & 63;
    st_bf16(sm + K2_OFF_A + r*144 + c*2, gv[r0*64 + e]);
  }
  for (int e = tid; e < 64*64; e += 256) {
    int k = e >> 6, n = e & 63;
    st_bf16(sm + K2_OFF_W + n*144 + k*2, w1[e]);
  }
  if (tid < 64) ((float*)(sm + K2_OFF_B))[tid] = b1[tid];
  __syncthreads();
  int wave = tid >> 5, lane = tid & 31, col16 = lane & 15, half = lane >> 4;
  for (int s = 0; s < 2; ++s) {
    int t0 = wave * 4 + 2 * s, t1 = t0 + 1;
    int m0 = (t0 >> 2) * 16, n0 = (t0 & 3) * 16;
    int m1 = (t1 >> 2) * 16, n1 = (t1 & 3) * 16;
    v8f acc0 = {0.f,0.f,0.f,0.f,0.f,0.f,0.f,0.f};
    v8f acc1 = {0.f,0.f,0.f,0.f,0.f,0.f,0.f,0.f};
#pragma unroll
    for (int kk = 0; kk < 64; kk += 32) {
      v16bf a0 = ld_frag(sm + K2_OFF_A, m0, 144, kk);
      v16bf b0 = ld_frag(sm + K2_OFF_W, n0, 144, kk);
      v16bf a1 = ld_frag(sm + K2_OFF_A, m1, 144, kk);
      v16bf b1f = ld_frag(sm + K2_OFF_W, n1, 144, kk);
      acc0 = WMMA_BF16(a0, b0, acc0);
      acc1 = WMMA_BF16(a1, b1f, acc1);
    }
#pragma unroll
    for (int q = 0; q < 2; ++q) {
      int mt = q ? m1 : m0, nt = q ? n1 : n0;
      v8f& acc = q ? acc1 : acc0;
      int col = nt + col16;
      float bs = ((float*)(sm + K2_OFF_B))[col];
#pragma unroll
      for (int v = 0; v < 8; ++v)
        tout[(r0 + mt + v + half*8)*64 + col] = (__bf16)fmaxf(acc[v] + bs, 0.f);
    }
  }
}

// ---- readout stage 2: out = t @ ro_w2 + b2, N permuted to output layout ----
// n' = c*900 + y*30 + x  maps to weight col j = (n'%900)*11 + n'/900,
// so stores d_out[b*9900 + n'] are fully coalesced.
#define B_OFF_A 0                  // 128*144
#define B_OFF_W 18432              // 256*144
#define B_OFF_B 55296              // 256 f32
#define SMEM_K2B 56320

__global__ __launch_bounds__(256) void readout_stage2(
    const unsigned short* __restrict__ tin, const float* __restrict__ w2,
    const float* __restrict__ b2, float* __restrict__ out) {
  extern __shared__ char sm[];
  int tid = (int)threadIdx.x;
  size_t m_base = (size_t)blockIdx.y * 128;
  int n_base = (int)blockIdx.x * 256;
  for (int e = tid; e < 128*64; e += 256) {
    int r = e >> 6, c = e & 63;
    *(unsigned short*)(sm + B_OFF_A + r*144 + c*2) = tin[m_base*64 + e];
  }
  {
    int np = n_base + tid;
    bool ok = np < OUTC;
    int j = ok ? ((np % 900) * 11 + np / 900) : 0;
    for (int k = 0; k < 64; ++k) {
      float w = ok ? w2[(size_t)k*OUTC + j] : 0.f;
      st_bf16(sm + B_OFF_W + tid*144 + k*2, w);
    }
    ((float*)(sm + B_OFF_B))[tid] = ok ? b2[j] : 0.f;
  }
  __syncthreads();
  int wave = tid >> 5, lane = tid & 31, col16 = lane & 15, half = lane >> 4;
  for (int s = 0; s < 8; ++s) {
    int t0 = wave * 16 + 2 * s, t1 = t0 + 1;
    int m0 = (t0 >> 4) * 16, n0 = (t0 & 15) * 16;
    int m1 = (t1 >> 4) * 16, n1 = (t1 & 15) * 16;
    v8f acc0 = {0.f,0.f,0.f,0.f,0.f,0.f,0.f,0.f};
    v8f acc1 = {0.f,0.f,0.f,0.f,0.f,0.f,0.f,0.f};
#pragma unroll
    for (int kk = 0; kk < 64; kk += 32) {
      v16bf a0 = ld_frag(sm + B_OFF_A, m0, 144, kk);
      v16bf b0 = ld_frag(sm + B_OFF_W, n0, 144, kk);
      v16bf a1 = ld_frag(sm + B_OFF_A, m1, 144, kk);
      v16bf b1f = ld_frag(sm + B_OFF_W, n1, 144, kk);
      acc0 = WMMA_BF16(a0, b0, acc0);
      acc1 = WMMA_BF16(a1, b1f, acc1);
    }
#pragma unroll
    for (int q = 0; q < 2; ++q) {
      int mt = q ? m1 : m0, nt = q ? n1 : n0;
      v8f& acc = q ? acc1 : acc0;
      int nc = nt + col16;
      int np = n_base + nc;
      if (np < OUTC) {
        float bs = ((float*)(sm + B_OFF_B))[nc];
#pragma unroll
        for (int v = 0; v < 8; ++v)
          out[(m_base + mt + v + half*8) * (size_t)OUTC + np] = acc[v] + bs;
      }
    }
  }
}

extern "C" void kernel_launch(void* const* d_in, const int* in_sizes, int n_in,
                              void* d_out, int out_size, void* d_ws, size_t ws_size,
                              hipStream_t stream) {
  (void)n_in; (void)out_size; (void)ws_size;
  const float* nf      = (const float*)d_in[0];
  const int*   nnodes  = (const int*)d_in[1];
  const float* embed_w = (const float*)d_in[2];
  const float* embed_b = (const float*)d_in[3];
  const float* edge_w1 = (const float*)d_in[4];
  const float* edge_b1 = (const float*)d_in[5];
  const float* edge_w2 = (const float*)d_in[6];
  const float* edge_b2 = (const float*)d_in[7];
  const float* node_w1 = (const float*)d_in[8];
  const float* node_b1 = (const float*)d_in[9];
  const float* node_w2 = (const float*)d_in[10];
  const float* node_b2 = (const float*)d_in[11];
  const float* ln_g    = (const float*)d_in[12];
  const float* ln_b    = (const float*)d_in[13];
  const float* ro_w1   = (const float*)d_in[14];
  const float* ro_b1   = (const float*)d_in[15];
  const float* ro_w2   = (const float*)d_in[16];
  const float* ro_b2   = (const float*)d_in[17];

  const int B = in_sizes[0] / (MAXO * NDIM);      // 16384
  float* gv = (float*)d_ws;
  __bf16* tbuf = (__bf16*)((char*)d_ws + (size_t)B * HDIM * sizeof(float));

  gnn_fused<<<B / GPB, 256, SMEM_K1, stream>>>(
      nf, nnodes, embed_w, embed_b,
      edge_w1, edge_b1, edge_w2, edge_b2,
      node_w1, node_b1, node_w2, node_b2,
      ln_g, ln_b, gv);

  readout_stage1<<<B / 128, 256, SMEM_K2A, stream>>>(gv, ro_w1, ro_b1, tbuf);

  dim3 g2((OUTC + 255) / 256, B / 128);
  readout_stage2<<<g2, 256, SMEM_K2B, stream>>>((const unsigned short*)tbuf, ro_w2, ro_b2,
                                                (float*)d_out);
}
